// Agent_3100966388056
// MI455X (gfx1250) — matrix operand order repaired and checked by reading, not compiled, not run on hardware
//
#include <hip/hip_runtime.h>
#include <hip/hip_bf16.h>

#define TT 128
#define BB 1024
#define TB (TT * BB)
#define OBS 63
#define HH 256
#define AA 6

typedef __attribute__((ext_vector_type(16))) __bf16 bf16x16;
typedef __attribute__((ext_vector_type(8)))  float  f32x8;

union ABFrag { bf16x16 v; unsigned u[8]; };

// ---- hardware tanh (gfx1250 v_tanh_f32) with libm fallback ----
#if defined(__has_builtin)
#if __has_builtin(__builtin_amdgcn_tanhf)
#define HW_TANH(x) __builtin_amdgcn_tanhf(x)
#endif
#endif
#ifndef HW_TANH
#define HW_TANH(x) tanhf(x)
#endif

__device__ inline float fast_tanh(float x) { return HW_TANH(x); }
// branchless sigmoid via hardware tanh: sigm(x) = 0.5 + 0.5*tanh(x/2)
__device__ inline float sigm(float x) { return fmaf(0.5f, HW_TANH(0.5f * x), 0.5f); }

// ---- WMMA fragment loaders (CDNA5 ISA 7.12.2 bf16 layouts, wave32) ----

// A fragment 16x32 (MxK) from row-major [M, ld] bf16.
// lane L: m = L&15, K-half = (L>>4)*8 ; VGPR v: k = ((v>>2)<<4) + (v&3)*2 (+half)
__device__ inline bf16x16 load_frag_a(const __bf16* src, int ld, int lane) {
    ABFrag f;
    int m  = lane & 15;
    int kh = (lane >> 4) << 3;
    const __bf16* p = src + m * ld + kh;
#pragma unroll
    for (int v = 0; v < 8; ++v) {
        int k = ((v >> 2) << 4) + ((v & 3) << 1);
        f.u[v] = *reinterpret_cast<const unsigned*>(p + k);
    }
    return f.v;
}

// B fragment 32x16 (KxN) from weight stored row-major [N, ldk] (i.e. B = W^T).
// lane L: n = L&15, K-half = (L>>4)*8 ; VGPR v: k = ((v>>2)<<4) + (v&3)*2 (+half)
__device__ inline bf16x16 load_frag_b(const __bf16* W, int ldk, int lane) {
    ABFrag f;
    int n  = lane & 15;
    int kb = (lane >> 4) << 3;
    const __bf16* p = W + n * ldk + kb;
#pragma unroll
    for (int v = 0; v < 8; ++v)
        f.u[v] = *reinterpret_cast<const unsigned*>(p + ((v >> 2) << 4) + ((v & 3) << 1));
    return f.v;
}

__device__ inline f32x8 wmma_bf16(bf16x16 a, bf16x16 b, f32x8 c) {
    return __builtin_amdgcn_wmma_f32_16x16x32_bf16(
        false, a, false, b, (short)0, c, false, false);
}

// ---------------- prep kernels ----------------

__global__ __launch_bounds__(256)
void pad_x_kernel(const float* __restrict__ x, __bf16* __restrict__ xb) {
    int i = blockIdx.x * 256 + threadIdx.x;          // over TB*64
    if (i >= TB * 64) return;
    int row = i >> 6, col = i & 63;
    xb[i] = (col < OBS) ? (__bf16)x[row * OBS + col] : (__bf16)0.0f;
}

__global__ __launch_bounds__(256)
void prep_w_kernel(const float* __restrict__ W1, const float* __restrict__ W2,
                   const float* __restrict__ Wih, const float* __restrict__ Whh,
                   const float* __restrict__ bih, const float* __restrict__ bhh,
                   __bf16* __restrict__ W1b, __bf16* __restrict__ W2b,
                   __bf16* __restrict__ Wcb, float* __restrict__ bcomb) {
    int i = blockIdx.x * 256 + threadIdx.x;          // up to 1024*512
    if (i < HH * 64) {
        int r = i >> 6, col = i & 63;
        W1b[i] = (col < OBS) ? (__bf16)W1[r * OBS + col] : (__bf16)0.0f;
    }
    if (i < HH * HH) W2b[i] = (__bf16)W2[i];
    if (i < 4 * HH * 2 * HH) {
        int r = i >> 9, col = i & 511;
        Wcb[i] = (col < HH) ? (__bf16)Wih[r * HH + col]
                            : (__bf16)Whh[r * HH + (col - HH)];
    }
    if (i < 4 * HH) bcomb[i] = bih[i] + bhh[i];
}

// ---------------- fused MLP: feat = tanh(tanh(x W1^T + b1) W2^T + b2) ----------------
// one 16-row tile per block, 8 waves, 2 n-tiles (of 16) per wave

__global__ __launch_bounds__(256)
void mlp_kernel(const __bf16* __restrict__ xb,    // [TB, 64]
                const __bf16* __restrict__ W1b,   // [256, 64]
                const float*  __restrict__ b1,
                const __bf16* __restrict__ W2b,   // [256, 256]
                const float*  __restrict__ b2,
                __bf16* __restrict__ feat)        // [TB, 256]
{
    __shared__ alignas(16) __bf16 h1[16 * 264];
    const int lane = threadIdx.x & 31;
    const int wave = threadIdx.x >> 5;
    const int row0 = blockIdx.x << 4;
    const int nt0 = wave * 2, nt1 = nt0 + 1;
    const int n_l  = lane & 15;
    const int moff = (lane >> 4) << 3;

    // layer 1 (K = 64)
    f32x8 a0 = {}, a1 = {};
#pragma unroll
    for (int kc = 0; kc < 64; kc += 32) {
        bf16x16 af = load_frag_a(xb + (size_t)row0 * 64 + kc, 64, lane);
        bf16x16 bf0 = load_frag_b(W1b + nt0 * 16 * 64 + kc, 64, lane);
        bf16x16 bf1 = load_frag_b(W1b + nt1 * 16 * 64 + kc, 64, lane);
        a0 = wmma_bf16(af, bf0, a0);
        a1 = wmma_bf16(af, bf1, a1);
    }
#pragma unroll
    for (int r = 0; r < 8; ++r) {
        int m = r + moff;
        int c0i = nt0 * 16 + n_l, c1i = nt1 * 16 + n_l;
        h1[m * 264 + c0i] = (__bf16)fast_tanh(a0[r] + b1[c0i]);
        h1[m * 264 + c1i] = (__bf16)fast_tanh(a1[r] + b1[c1i]);
    }
    __syncthreads();

    // layer 2 (K = 256)
    f32x8 d0 = {}, d1 = {};
#pragma unroll
    for (int kc = 0; kc < 256; kc += 32) {
        bf16x16 af = load_frag_a(h1 + kc, 264, lane);
        bf16x16 bf0 = load_frag_b(W2b + nt0 * 16 * 256 + kc, 256, lane);
        bf16x16 bf1 = load_frag_b(W2b + nt1 * 16 * 256 + kc, 256, lane);
        d0 = wmma_bf16(af, bf0, d0);
        d1 = wmma_bf16(af, bf1, d1);
    }
#pragma unroll
    for (int r = 0; r < 8; ++r) {
        int m = r + moff;
        int c0i = nt0 * 16 + n_l, c1i = nt1 * 16 + n_l;
        feat[(size_t)(row0 + m) * HH + c0i] = (__bf16)fast_tanh(d0[r] + b2[c0i]);
        feat[(size_t)(row0 + m) * HH + c1i] = (__bf16)fast_tanh(d1[r] + b2[c1i]);
    }
}

// ---------------- LSTM scan: 64 blocks x 16 batch rows, loop over T ----------------
// A-tile [16, 512] = [feat_t | h_t] in LDS; g = A @ Wcomb^T (K=512), c in registers.
// LDS row stride 520 bf16 = 1040 B (16B multiple -> b128 staging legal).

__global__ __launch_bounds__(256)
void lstm_kernel(const __bf16* __restrict__ feat,  // [T, B, 256]
                 const int*    __restrict__ done,  // [T*B]
                 const float*  __restrict__ h0,    // [B, 256]
                 const float*  __restrict__ c0,
                 const __bf16* __restrict__ Wcb,   // [1024, 512]
                 const float*  __restrict__ bcomb, // [1024]
                 __bf16* __restrict__ hs,          // [T*B, 256] bf16
                 float* __restrict__ hT,           // [B, 256]
                 float* __restrict__ cT)           // [B, 256]
{
    __shared__ alignas(16) __bf16 ldsA[16 * 520]; // cols 0..255 feat_t, 256..511 h
    __shared__ float  bias_s[4 * HH];
    __shared__ float  mask_s[TT * 16];     // (1 - done) per step / row
    const int lane = threadIdx.x & 31;
    const int wave = threadIdx.x >> 5;
    const int tid  = threadIdx.x;
    const int r0   = blockIdx.x << 4;

    for (int i = tid; i < 4 * HH; i += 256) bias_s[i] = bcomb[i];
    for (int i = tid; i < TT * 16; i += 256) {
        int t = i >> 4, m = i & 15;
        mask_s[i] = 1.0f - (float)done[t * BB + r0 + m];
    }

    const int n_l  = lane & 15;
    const int moff = (lane >> 4) << 3;
    const int jt0 = wave * 2, jt1 = jt0 + 1;
    const int j0 = jt0 * 16 + n_l, j1 = jt1 * 16 + n_l;

    __syncthreads();   // masks ready

    // initial state, masked by done[0]
    float cr0[8], cr1[8];
#pragma unroll
    for (int r = 0; r < 8; ++r) {
        int m = r + moff;
        float mk = mask_s[m];
        ldsA[m * 520 + 256 + j0] = (__bf16)(h0[(r0 + m) * HH + j0] * mk);
        ldsA[m * 520 + 256 + j1] = (__bf16)(h0[(r0 + m) * HH + j1] * mk);
        cr0[r] = c0[(r0 + m) * HH + j0] * mk;
        cr1[r] = c0[(r0 + m) * HH + j1] * mk;
    }

    const float bi0 = bias_s[0 * HH + j0], bff0 = bias_s[1 * HH + j0],
                bg0 = bias_s[2 * HH + j0], bo0  = bias_s[3 * HH + j0];
    const float bi1 = bias_s[0 * HH + j1], bff1 = bias_s[1 * HH + j1],
                bg1 = bias_s[2 * HH + j1], bo1  = bias_s[3 * HH + j1];

    for (int t = 0; t < TT; ++t) {
        // stage 1: stream feat_t tile into LDS at b128 granularity
        // (16 rows x 512 B; 512 uint4 total -> 2 per thread)
        {
            const uint4* ft =
                (const uint4*)(feat + ((size_t)t * BB + r0) * HH); // 32 uint4/row
#pragma unroll
            for (int i = tid; i < 512; i += 256) {
                int m = i >> 5, q = i & 31;
                *reinterpret_cast<uint4*>(&ldsA[m * 520 + q * 8]) = ft[m * 32 + q];
            }
        }
        // prefetch next step's feat tile (8 KB = 64 x 128B lines) so its
        // latency hides under this step's WMMA chain (global_prefetch_b8)
        if (t + 1 < TT) {
            const char* fn =
                (const char*)(feat + ((size_t)(t + 1) * BB + r0) * HH);
            if (tid < 64) __builtin_prefetch(fn + tid * 128, 0, 3);
        }
        __syncthreads();

        // stage 2: gates GEMM, 8 output tiles per wave (4 gates x 2 j-tiles), K=512
        f32x8 acc[4][2] = {};
        for (int kc = 0; kc < 512; kc += 32) {
            bf16x16 af = load_frag_a(ldsA + kc, 520, lane);
#pragma unroll
            for (int g = 0; g < 4; ++g) {
                bf16x16 b0 = load_frag_b(
                    Wcb + (size_t)(g * HH + jt0 * 16) * 512 + kc, 512, lane);
                bf16x16 b1 = load_frag_b(
                    Wcb + (size_t)(g * HH + jt1 * 16) * 512 + kc, 512, lane);
                acc[g][0] = wmma_bf16(af, b0, acc[g][0]);
                acc[g][1] = wmma_bf16(af, b1, acc[g][1]);
            }
        }
        __syncthreads();   // all reads of ldsA complete

        // stage 3: elementwise LSTM cell (hardware tanh), write h back
        // (pre-masked with done[t+1])
#pragma unroll
        for (int r = 0; r < 8; ++r) {
            int m = r + moff;
            float mnext = (t + 1 < TT) ? mask_s[(t + 1) * 16 + m] : 1.0f;
            {
                float ig = sigm(acc[0][0][r] + bi0);
                float fg = sigm(acc[1][0][r] + bff0);
                float gg = fast_tanh(acc[2][0][r] + bg0);
                float og = sigm(acc[3][0][r] + bo0);
                float cc = fg * cr0[r] + ig * gg;
                float hh = og * fast_tanh(cc);
                hs[((size_t)t * BB + r0 + m) * HH + j0] = (__bf16)hh;
                if (t == TT - 1) {
                    hT[(r0 + m) * HH + j0] = hh;
                    cT[(r0 + m) * HH + j0] = cc;
                }
                cr0[r] = cc * mnext;
                ldsA[m * 520 + 256 + j0] = (__bf16)(hh * mnext);
            }
            {
                float ig = sigm(acc[0][1][r] + bi1);
                float fg = sigm(acc[1][1][r] + bff1);
                float gg = fast_tanh(acc[2][1][r] + bg1);
                float og = sigm(acc[3][1][r] + bo1);
                float cc = fg * cr1[r] + ig * gg;
                float hh = og * fast_tanh(cc);
                hs[((size_t)t * BB + r0 + m) * HH + j1] = (__bf16)hh;
                if (t == TT - 1) {
                    hT[(r0 + m) * HH + j1] = hh;
                    cT[(r0 + m) * HH + j1] = cc;
                }
                cr1[r] = cc * mnext;
                ldsA[m * 520 + 256 + j1] = (__bf16)(hh * mnext);
            }
        }
        // next iteration's stage-1 only touches feat region; stage-2 barrier
        // guarantees ordering of h writes vs. GEMM reads.
    }
}

// ---------------- heads: out[:,0:6] = hs Wa^T + ba ; out[:,6] = hs Wc^T + bc ----------------

__global__ __launch_bounds__(256)
void head_kernel(const __bf16* __restrict__ hs, const float* __restrict__ Wa,
                 const float* __restrict__ ba, const float* __restrict__ Wc,
                 const float* __restrict__ bc, float* __restrict__ out) {
    int gid = blockIdx.x * 256 + threadIdx.x;
    int row = gid >> 3, c = gid & 7;
    if (row >= TB || c >= 7) return;
    const __bf16* h = hs + (size_t)row * HH;
    const float* w = (c < AA) ? (Wa + c * HH) : Wc;
    float acc = (c < AA) ? ba[c] : bc[0];
#pragma unroll 8
    for (int k = 0; k < HH; ++k) acc += (float)h[k] * w[k];
    out[(size_t)row * (AA + 1) + c] = acc;
}

// ---------------- launcher ----------------

extern "C" void kernel_launch(void* const* d_in, const int* in_sizes, int n_in,
                              void* d_out, int out_size, void* d_ws, size_t ws_size,
                              hipStream_t stream) {
    (void)in_sizes; (void)n_in; (void)out_size; (void)ws_size;
    const float* x    = (const float*)d_in[0];
    const int*   done = (const int*)  d_in[1];
    const float* h0   = (const float*)d_in[2];
    const float* c0   = (const float*)d_in[3];
    const float* W1   = (const float*)d_in[4];
    const float* b1   = (const float*)d_in[5];
    const float* W2   = (const float*)d_in[6];
    const float* b2   = (const float*)d_in[7];
    const float* Wih  = (const float*)d_in[8];
    const float* bih  = (const float*)d_in[9];
    const float* Whh  = (const float*)d_in[10];
    const float* bhh  = (const float*)d_in[11];
    const float* Wa   = (const float*)d_in[12];
    const float* ba   = (const float*)d_in[13];
    const float* Wc   = (const float*)d_in[14];
    const float* bc   = (const float*)d_in[15];
    float* out = (float*)d_out;
    float* hT  = out + (size_t)TB * (AA + 1);
    float* cT  = hT + (size_t)BB * HH;

    char* ws = (char*)d_ws;
    size_t off = 0;
    auto alloc = [&](size_t bytes) -> void* {
        off = (off + 255) & ~(size_t)255;
        void* p = ws + off;
        off += bytes;
        return p;
    };
    __bf16* xb    = (__bf16*)alloc((size_t)TB * 64 * 2);        // 16 MB
    __bf16* W1b   = (__bf16*)alloc((size_t)HH * 64 * 2);
    __bf16* W2b   = (__bf16*)alloc((size_t)HH * HH * 2);
    __bf16* Wcb   = (__bf16*)alloc((size_t)4 * HH * 2 * HH * 2); // 1 MB
    float*  bcomb = (float*) alloc((size_t)4 * HH * 4);
    __bf16* feat  = (__bf16*)alloc((size_t)TB * HH * 2);        // 64 MB
    __bf16* hsb   = (__bf16*)alloc((size_t)TB * HH * 2);        // 64 MB

    pad_x_kernel<<<(TB * 64 + 255) / 256, 256, 0, stream>>>(x, xb);
    prep_w_kernel<<<(4 * HH * 2 * HH + 255) / 256, 256, 0, stream>>>(
        W1, W2, Wih, Whh, bih, bhh, W1b, W2b, Wcb, bcomb);
    mlp_kernel<<<TB / 16, 256, 0, stream>>>(xb, W1b, b1, W2b, b2, feat);
    lstm_kernel<<<BB / 16, 256, 0, stream>>>(feat, done, h0, c0, Wcb, bcomb,
                                             hsb, hT, cT);
    head_kernel<<<(TB * 8 + 255) / 256, 256, 0, stream>>>(hsb, Wa, ba, Wc, bc, out);
}
